// MiniBatchDiscrimination_43508018708960
// MI455X (gfx1250) — matrix operand order repaired and checked by reading, not compiled, not run on hardware
//
#include <hip/hip_runtime.h>

typedef __attribute__((ext_vector_type(16))) __bf16 v16bf;
typedef __attribute__((ext_vector_type(8)))  float  v8f;

#define N_   32
#define HW_  256    // H*W
#define F_   256
#define B_   64
#define C_   8
#define COLS 512    // B_*C_
#define OUTC 320    // F_+B_

// ---------------------------------------------------------------------------
// Prepack T [256][512] f32 -> bf16 in WMMA-B-fragment lane order:
// Tpk[((ct*8 + ks)*32 + lane)*16 + i] = bf16( T[(ks*32 + (lane/16)*16 + i)*512 + ct*16 + lane%16] )
// so the main kernel's B fragment is one contiguous 32B load per lane.
// ---------------------------------------------------------------------------
__global__ void prepackT_kernel(const float* __restrict__ T, __bf16* __restrict__ Tpk) {
    int idx  = blockIdx.x * blockDim.x + threadIdx.x;   // 0 .. 131071
    int i    = idx & 15;
    int lane = (idx >> 4) & 31;
    int ks   = (idx >> 9) & 7;
    int ct   = idx >> 12;                               // 0..31
    int krow = ks * 32 + (lane >> 4) * 16 + i;
    int col  = ct * 16 + (lane & 15);
    Tpk[idx] = (__bf16)T[krow * COLS + col];
}

// ---------------------------------------------------------------------------
// Fused: per-(h,w) WG: x passthrough + bf16 WMMA GEMM into LDS + pairwise
// L1/exp reduction, all in one pass.
// ---------------------------------------------------------------------------
__global__ __launch_bounds__(256)
void mbd_kernel(const float* __restrict__ x, const __bf16* __restrict__ Tpk,
                float* __restrict__ out) {
    __shared__ __bf16 Xs[N_][264];   // padded: A-frag reads conflict-free
    __shared__ float  Ms[N_][516];   // padded: D stores / C reads conflict-free

    const int hw   = blockIdx.x;     // h*16 + w
    const int t    = threadIdx.x;    // 0..255
    const int lane = t & 31;
    const int wave = t >> 5;         // 0..7

    // ---- Stage A: load x rows for this pixel, passthrough, bf16 -> LDS ----
    // x[n,h,w,f] at (n*HW_ + hw)*F_ + f ; out[n,h,w,f] at (n*HW_ + hw)*OUTC + f
    for (int n = 0; n < N_; ++n) {
        float v = x[(n * HW_ + hw) * F_ + t];
        out[(n * HW_ + hw) * OUTC + t] = v;
        Xs[n][t] = (__bf16)v;
    }
    __syncthreads();

    // ---- Stage B: Ms[32][512] = Xs[32][256] * T[256][512] via bf16 WMMA ---
    {
        const int m    = lane & 15;
        const int half = lane >> 4;
        v8f acc[2][4];
        const v8f vzero = {};
        #pragma unroll
        for (int nt = 0; nt < 2; ++nt)
            #pragma unroll
            for (int ct = 0; ct < 4; ++ct)
                acc[nt][ct] = vzero;

        for (int ks = 0; ks < 8; ++ks) {
            const int k0 = ks * 32;
            // A fragments (contiguous K, matching prepacked B): half 0 -> K {0..15},
            // half 1 -> K {16..31}, row M = lane%16 (+16 for 2nd n-tile).
            v16bf a0, a1;
            {
                const __bf16* p0 = &Xs[m     ][k0 +     half * 16];
                const __bf16* q0 = &Xs[m     ][k0 + 8 + half * 16];
                const __bf16* p1 = &Xs[16 + m][k0 +     half * 16];
                const __bf16* q1 = &Xs[16 + m][k0 + 8 + half * 16];
                #pragma unroll
                for (int i = 0; i < 8; ++i) {
                    a0[i] = p0[i]; a0[8 + i] = q0[i];
                    a1[i] = p1[i]; a1[8 + i] = q1[i];
                }
            }
            #pragma unroll
            for (int ct = 0; ct < 4; ++ct) {
                const int ctg = wave * 4 + ct;   // global col-tile 0..31
                v16bf b = *(const v16bf*)&Tpk[(((ctg * 8 + ks) * 32) + lane) * 16];
                acc[0][ct] = __builtin_amdgcn_wmma_f32_16x16x32_bf16(
                    false, a0, false, b, (short)0, acc[0][ct], false, false);
                acc[1][ct] = __builtin_amdgcn_wmma_f32_16x16x32_bf16(
                    false, a1, false, b, (short)0, acc[1][ct], false, false);
            }
        }

        // D layout: lanes 0-15 col N=lane, rows r ; lanes 16-31 rows 8+r.
        #pragma unroll
        for (int nt = 0; nt < 2; ++nt) {
            #pragma unroll
            for (int ct = 0; ct < 4; ++ct) {
                const int colb = (wave * 4 + ct) * 16 + (lane & 15);
                const int mrow = nt * 16 + (lane >> 4) * 8;
                #pragma unroll
                for (int r = 0; r < 8; ++r)
                    Ms[mrow + r][colb] = acc[nt][ct][r];
            }
        }
    }
    __syncthreads();

    // ---- Stage C: o[n,b] = sum_j exp(-sum_c |M[n,b,c]-M[j,b,c]|) ----------
    // Thread handles 8 (b,n) outputs; a whole wave shares one b per iteration
    // so all Ms[j][b*8+c] reads are LDS broadcasts.
    for (int it = 0; it < 8; ++it) {
        const int idx = t + 256 * it;   // 0..2047
        const int b   = idx >> 5;       // 0..63
        const int n   = idx & 31;
        const float* mi = &Ms[n][b * C_];
        float m0 = mi[0], m1 = mi[1], m2 = mi[2], m3 = mi[3];
        float m4 = mi[4], m5 = mi[5], m6 = mi[6], m7 = mi[7];
        float o = 0.0f;
        for (int j = 0; j < N_; ++j) {
            const float* mj = &Ms[j][b * C_];
            float d = fabsf(m0 - mj[0]) + fabsf(m1 - mj[1])
                    + fabsf(m2 - mj[2]) + fabsf(m3 - mj[3])
                    + fabsf(m4 - mj[4]) + fabsf(m5 - mj[5])
                    + fabsf(m6 - mj[6]) + fabsf(m7 - mj[7]);
            o += expf(-d);
        }
        out[(n * HW_ + hw) * OUTC + F_ + b] = o;
    }
}

extern "C" void kernel_launch(void* const* d_in, const int* in_sizes, int n_in,
                              void* d_out, int out_size, void* d_ws, size_t ws_size,
                              hipStream_t stream) {
    const float* x = (const float*)d_in[0];   // [32,16,16,256] f32
    const float* T = (const float*)d_in[1];   // [256,64,8]     f32
    float* out = (float*)d_out;               // [32,16,16,320] f32
    __bf16* Tpk = (__bf16*)d_ws;              // 256 KB prepacked B fragments

    prepackT_kernel<<<512, 256, 0, stream>>>(T, Tpk);
    mbd_kernel<<<HW_, 256, 0, stream>>>(x, Tpk, out);
}
